// NATTEN_Encoder_20761871909032
// MI455X (gfx1250) — matrix operand order repaired and checked by reading, not compiled, not run on hardware
//
#include <hip/hip_runtime.h>
#include <hip/hip_bf16.h>

// ---------------------------------------------------------------------------
// NATTEN-style 2D neighborhood attention encoder for gfx1250 (MI455X).
// B=1, H=64, W=352 -> 176 -> 88, C=24, 4 heads, head_dim=6.
//
// Dense projections use v_wmma_f32_16x16x32_f16. All operand feeds are
// branch-free: weights are pre-padded to f16 [32][80]/[32][32] tiles in
// workspace, activations live in a zero-padded stride-32 f32 layout, and the
// qkv tensor uses stride 80 — so A is 4x global_load_b128 per lane and B is
// 16 strided f16 loads per lane, with unconditional stores.
// ---------------------------------------------------------------------------

typedef __attribute__((ext_vector_type(16))) _Float16 v16h;
typedef __attribute__((ext_vector_type(8)))  float    v8f;

#define C_DIM    24
#define C_PAD    32     // padded activation stride
#define NHEAD    4
#define HDIM     6
#define QKV_N    72
#define QKV_PAD  80     // padded qkv stride
#define WAVES_PER_BLOCK 8

// ---------------------------------------------------------------------------
// One-time prep: pad weights to f16 and biases to f32 in workspace.
// ---------------------------------------------------------------------------
__global__ __launch_bounds__(256)
void k_prep(const float* __restrict__ qkv_w, const float* __restrict__ qkv_b,
            const float* __restrict__ out_w, const float* __restrict__ out_b,
            _Float16* __restrict__ wq16, _Float16* __restrict__ wo16,
            float* __restrict__ qb, float* __restrict__ ob)
{
    int t = blockIdx.x * blockDim.x + threadIdx.x;
    if (t < 4 * 32 * QKV_PAD) {
        int layer = t / (32 * QKV_PAD);
        int r     = (t / QKV_PAD) % 32;
        int n     = t % QKV_PAD;
        wq16[t] = (r < C_DIM && n < QKV_N)
                      ? (_Float16)qkv_w[(layer * C_DIM + r) * QKV_N + n]
                      : (_Float16)0.f;
    }
    if (t < 4 * 32 * C_PAD) {
        int layer = t / (32 * C_PAD);
        int r     = (t / C_PAD) % 32;
        int n     = t % C_PAD;
        wo16[t] = (r < C_DIM && n < C_DIM)
                      ? (_Float16)out_w[(layer * C_DIM + r) * C_DIM + n]
                      : (_Float16)0.f;
    }
    if (t < 4 * QKV_PAD) {
        int layer = t / QKV_PAD, n = t % QKV_PAD;
        qb[t] = (n < QKV_N) ? qkv_b[layer * QKV_N + n] : 0.f;
    }
    if (t < 4 * C_PAD) {
        int layer = t / C_PAD, n = t % C_PAD;
        ob[t] = (n < C_DIM) ? out_b[layer * C_DIM + n] : 0.f;
    }
}

__global__ __launch_bounds__(256)
void k_import(const float* __restrict__ x, float* __restrict__ x0, int P)
{
    int t = blockIdx.x * blockDim.x + threadIdx.x;
    if (t >= P * C_PAD) return;
    int c = t & (C_PAD - 1);
    int p = t >> 5;
    x0[t] = (c < C_DIM) ? x[p * C_DIM + c] : 0.f;
}

// 16x32 f16 A tile from padded stride-32 rows; four unconditional b128 loads.
__device__ __forceinline__ v16h load_A(const float* __restrict__ act, int mrow, int lane)
{
    const int hiK = (lane >> 4) * 8;                  // 0 or 8
    const float* rowp = act + (size_t)mrow * C_PAD;
    float4 a0 = *(const float4*)(rowp + hiK);
    float4 a1 = *(const float4*)(rowp + hiK + 4);
    float4 a2 = *(const float4*)(rowp + 16 + hiK);
    float4 a3 = *(const float4*)(rowp + 20 + hiK);
    v16h a;
    a[0]  = (_Float16)a0.x; a[1]  = (_Float16)a0.y; a[2]  = (_Float16)a0.z; a[3]  = (_Float16)a0.w;
    a[4]  = (_Float16)a1.x; a[5]  = (_Float16)a1.y; a[6]  = (_Float16)a1.z; a[7]  = (_Float16)a1.w;
    a[8]  = (_Float16)a2.x; a[9]  = (_Float16)a2.y; a[10] = (_Float16)a2.z; a[11] = (_Float16)a2.w;
    a[12] = (_Float16)a3.x; a[13] = (_Float16)a3.y; a[14] = (_Float16)a3.z; a[15] = (_Float16)a3.w;
    return a;
}

// qkv[P][80] = x[P][32] @ wq16[32][80] + qb[80]; 5 WMMAs per wave.
__global__ __launch_bounds__(32 * WAVES_PER_BLOCK)
void k_gemm_qkv(const float* __restrict__ x, const _Float16* __restrict__ wq16,
                const float* __restrict__ qb, float* __restrict__ qkv, int P)
{
    const int lane = threadIdx.x & 31;
    const int wave = threadIdx.x >> 5;
    const int tile = blockIdx.x * WAVES_PER_BLOCK + wave;
    if (tile * 16 >= P) return;

    const v16h a = load_A(x, tile * 16 + (lane & 15), lane);
    const int mbase = tile * 16 + ((lane >> 4) << 3);
    const int ncol  = lane & 15;

#pragma unroll
    for (int t = 0; t < 5; ++t) {
        v16h bm;
#pragma unroll
        for (int s = 0; s < 16; ++s) {
            int kk = (s < 8) ? ((lane >> 4) * 8 + s) : (16 + (lane >> 4) * 8 + (s - 8));
            bm[s] = wq16[(size_t)kk * QKV_PAD + t * 16 + (lane & 15)];
        }
        v8f c = {};
        c = __builtin_amdgcn_wmma_f32_16x16x32_f16(
                false, a, false, bm, (short)0, c, false, false);
        int n = t * 16 + ncol;
        float bias = qb[n];
#pragma unroll
        for (int r = 0; r < 8; ++r)
            qkv[(size_t)(mbase + r) * QKV_PAD + n] = c[r] + bias;
    }
}

// y[P][32] = att[P][32] @ wo16[32][32] + ob[32] + res[P][32]; 2 WMMAs per wave.
__global__ __launch_bounds__(32 * WAVES_PER_BLOCK)
void k_gemm_out(const float* __restrict__ att, const _Float16* __restrict__ wo16,
                const float* __restrict__ ob, const float* __restrict__ res,
                float* __restrict__ y, int P)
{
    const int lane = threadIdx.x & 31;
    const int wave = threadIdx.x >> 5;
    const int tile = blockIdx.x * WAVES_PER_BLOCK + wave;
    if (tile * 16 >= P) return;

    const v16h a = load_A(att, tile * 16 + (lane & 15), lane);
    const int mbase = tile * 16 + ((lane >> 4) << 3);
    const int ncol  = lane & 15;

#pragma unroll
    for (int t = 0; t < 2; ++t) {
        v16h bm;
#pragma unroll
        for (int s = 0; s < 16; ++s) {
            int kk = (s < 8) ? ((lane >> 4) * 8 + s) : (16 + (lane >> 4) * 8 + (s - 8));
            bm[s] = wo16[(size_t)kk * C_PAD + t * 16 + (lane & 15)];
        }
        v8f c = {};
        c = __builtin_amdgcn_wmma_f32_16x16x32_f16(
                false, a, false, bm, (short)0, c, false, false);
        int n = t * 16 + ncol;
        float bias = ob[n];
#pragma unroll
        for (int r = 0; r < 8; ++r) {
            size_t idx = (size_t)(mbase + r) * C_PAD + n;
            y[idx] = c[r] + bias + res[idx];
        }
    }
}

// Neighborhood attention with clamped window start; online softmax.
__global__ __launch_bounds__(256)
void k_natten(const float* __restrict__ qkv, float* __restrict__ att_out,
              int H, int W, int win)
{
    int t = blockIdx.x * blockDim.x + threadIdx.x;
    int P = H * W;
    if (t >= P * NHEAD) return;
    int head = t & 3;
    int p    = t >> 2;
    int h = p / W, w = p % W;

    int half = win >> 1;
    int sh = h - half; sh = sh < 0 ? 0 : sh; { int mx = H - win; sh = sh > mx ? mx : sh; }
    int sw = w - half; sw = sw < 0 ? 0 : sw; { int mx = W - win; sw = sw > mx ? mx : sw; }

    const float scale = 0.4082482904638630f;          // 1/sqrt(6)
    float q[HDIM];
    const float* qp = qkv + (size_t)p * QKV_PAD + head * HDIM;
#pragma unroll
    for (int d = 0; d < HDIM; ++d) q[d] = qp[d] * scale;

    float m = -1e30f, l = 0.f;
    float acc[HDIM];
#pragma unroll
    for (int d = 0; d < HDIM; ++d) acc[d] = 0.f;

    for (int i = 0; i < win; ++i) {
        const float* krow = qkv + ((size_t)((sh + i) * W + sw)) * QKV_PAD
                                + C_DIM + head * HDIM;
        for (int j = 0; j < win; ++j) {
            const float* kp = krow + j * QKV_PAD;
            float s = 0.f;
#pragma unroll
            for (int d = 0; d < HDIM; ++d) s = fmaf(q[d], kp[d], s);
            const float* vp = kp + C_DIM;
            float mn = fmaxf(m, s);
            float cf = __expf(m - mn);
            float pf = __expf(s - mn);
            l = l * cf + pf;
#pragma unroll
            for (int d = 0; d < HDIM; ++d) acc[d] = fmaf(acc[d], cf, pf * vp[d]);
            m = mn;
        }
    }
    float inv = 1.f / l;
    float* o = att_out + (size_t)p * C_PAD + head * HDIM;
#pragma unroll
    for (int d = 0; d < HDIM; ++d) o[d] = acc[d] * inv;
    if (head == 0) {                                  // keep pad cols zero
        float4 z = {0.f, 0.f, 0.f, 0.f};
        *(float4*)(att_out + (size_t)p * C_PAD + 24) = z;
        *(float4*)(att_out + (size_t)p * C_PAD + 28) = z;
    }
}

// Width-halving max pool on padded rows.
__global__ __launch_bounds__(256)
void k_pool(const float* __restrict__ in, float* __restrict__ out,
            int H, int Wout)
{
    int t = blockIdx.x * blockDim.x + threadIdx.x;
    int n = H * Wout * C_PAD;
    if (t >= n) return;
    int c  = t & (C_PAD - 1);
    int wo = (t >> 5) % Wout;
    int h  = t / (C_PAD * Wout);
    const float* a = in + ((size_t)(h * Wout * 2 + wo * 2)) * C_PAD + c;
    out[t] = fmaxf(a[0], a[C_PAD]);
}

// Final transpose: [H][W][32] (24 used) -> [H][24][W]
__global__ __launch_bounds__(256)
void k_transpose(const float* __restrict__ in, float* __restrict__ out,
                 int H, int W)
{
    int t = blockIdx.x * blockDim.x + threadIdx.x;
    int n = H * W * C_DIM;
    if (t >= n) return;
    int w = t % W;
    int c = (t / W) % C_DIM;
    int h = t / (W * C_DIM);
    out[t] = in[((size_t)(h * W + w)) * C_PAD + c];
}

// ---------------------------------------------------------------------------
extern "C" void kernel_launch(void* const* d_in, const int* in_sizes, int n_in,
                              void* d_out, int out_size, void* d_ws, size_t ws_size,
                              hipStream_t stream)
{
    const float* x_in  = (const float*)d_in[0];
    const float* qkv_w = (const float*)d_in[1];
    const float* qkv_b = (const float*)d_in[2];
    const float* out_w = (const float*)d_in[3];
    const float* out_b = (const float*)d_in[4];

    const int H  = 64;
    const int P0 = H * 352;

    float* ws   = (float*)d_ws;
    float* xA   = ws;                                    // P0*32
    float* xB   = xA  + (size_t)P0 * C_PAD;              // P0*32
    float* att  = xB  + (size_t)P0 * C_PAD;              // P0*32
    float* qkv  = att + (size_t)P0 * C_PAD;              // P0*80
    float* qb   = qkv + (size_t)P0 * QKV_PAD;            // 4*80
    float* ob   = qb  + 4 * QKV_PAD;                     // 4*32
    _Float16* wq16 = (_Float16*)(ob + 4 * C_PAD);        // 4*32*80 (32B-aligned)
    _Float16* wo16 = wq16 + 4 * 32 * QKV_PAD;            // 4*32*32

    k_prep<<<(4 * 32 * QKV_PAD + 255) / 256, 256, 0, stream>>>(
        qkv_w, qkv_b, out_w, out_b, wq16, wo16, qb, ob);
    k_import<<<(P0 * C_PAD + 255) / 256, 256, 0, stream>>>(x_in, xA, P0);

    const int wins[4] = {5, 5, 25, 25};
    int W = 352;
    const float* cur = xA;

    for (int layer = 0; layer < 4; ++layer) {
        int P      = H * W;
        int gemmB  = (P / 16 + WAVES_PER_BLOCK - 1) / WAVES_PER_BLOCK;

        k_gemm_qkv<<<gemmB, 32 * WAVES_PER_BLOCK, 0, stream>>>(
            cur, wq16 + (size_t)layer * 32 * QKV_PAD, qb + layer * QKV_PAD, qkv, P);

        k_natten<<<(P * NHEAD + 255) / 256, 256, 0, stream>>>(
            qkv, att, H, W, wins[layer]);

        // y must differ from cur; pooled result must differ from y.
        float* y = (cur == xA) ? xB : xA;
        k_gemm_out<<<gemmB, 32 * WAVES_PER_BLOCK, 0, stream>>>(
            att, wo16 + (size_t)layer * 32 * C_PAD, ob + layer * C_PAD, cur, y, P);

        if (layer < 2) {
            W /= 2;
            float* pooled = (y == xA) ? xB : xA;
            k_pool<<<(H * W * C_PAD + 255) / 256, 256, 0, stream>>>(y, pooled, H, W);
            cur = pooled;
        } else {
            cur = y;
        }
    }
    // L0: cur xA -> y xB -> pool xA ; L1: cur xA -> y xB -> pool xA
    // L2: cur xA -> y xB ; L3: cur xB -> y xA  => final in xA

    k_transpose<<<(H * 88 * C_DIM + 255) / 256, 256, 0, stream>>>(
        cur, (float*)d_out, H, 88);
}
